// DPLLM_Linear_65850438582285
// MI455X (gfx1250) — compile-verified, matrix operand order checked by reading, not run on hardware
//
#include <hip/hip_runtime.h>

// ---------------------------------------------------------------------------
// DP-LLM any-precision linear for MI455X (gfx1250, wave32, WMMA + async-LDS).
//   Pass 1: bitplane LUT dequant -> f16 W_lo / W_hi in workspace (2x33.5 MB).
//   Pass 2: cast x f32 -> f16 in workspace (67 MB; lives in 192 MB L2).
//   Pass 3: per-token ||x|| > 64 mask (8 KB).
//   Pass 4: dual-accumulate WMMA GEMM. All operands f16, staged with
//           GLOBAL_LOAD_ASYNC_TO_LDS_B128 into double-buffered LDS
//           (ASYNCcnt-tracked), 16x v_wmma_f32_16x16x32_f16 per K-step,
//           epilogue does per-token hi/lo select + bias.
// ---------------------------------------------------------------------------

typedef __attribute__((ext_vector_type(16))) _Float16 v16h;
typedef __attribute__((ext_vector_type(8)))  float    v8f;

#define IN_F   4096
#define OUT_F  4096
#define BATCH  4
#define SEQ    2048
#define M_TOT  (BATCH * SEQ)      // 8192 tokens
#define I32    (IN_F / 32)        // 128 packed words per output row per plane

#define BM 128
#define BN 128
#define BK 32

#define W_BYTES  ((size_t)OUT_F * IN_F * sizeof(_Float16))   // 33554432
#define XH_BYTES ((size_t)M_TOT * IN_F * sizeof(_Float16))   // 67108864

// ------------------- gfx1250 async global->LDS copy (16B/lane) -------------
__device__ __forceinline__ void async_copy_b128(const _Float16* gsrc,
                                                _Float16* ldst) {
  unsigned lds_addr = (unsigned)(unsigned long long)(uintptr_t)ldst;
  unsigned long long gaddr = (unsigned long long)(uintptr_t)gsrc;
  asm volatile("global_load_async_to_lds_b128 %0, %1, off"
               :: "v"(lds_addr), "v"(gaddr) : "memory");
}

__device__ __forceinline__ void wait_async_zero() {
  asm volatile("s_wait_asynccnt 0" ::: "memory");
}

// -------------------------- Pass 1: dequant --------------------------------
__global__ __launch_bounds__(256)
void dequant_kbit_kernel(const int* __restrict__ q,
                         const float* __restrict__ lut4,
                         const float* __restrict__ lut8,
                         _Float16* __restrict__ wl,
                         _Float16* __restrict__ wh) {
  int gid = blockIdx.x * 256 + threadIdx.x;      // 0 .. OUT_F*I32-1 (exact)
  int o = gid >> 7;                              // /128
  int w = gid & 127;

  unsigned p[8];
#pragma unroll
  for (int b = 0; b < 8; ++b)
    p[b] = (unsigned)q[(size_t)b * OUT_F * I32 + gid];

  const float* l4 = lut4 + (size_t)o * 16;
  const float* l8 = lut8 + (size_t)o * 256;
  _Float16* ol = wl + (size_t)o * IN_F + w * 32;
  _Float16* oh = wh + (size_t)o * IN_F + w * 32;

#pragma unroll
  for (int j = 0; j < 32; ++j) {
    unsigned i4 =  ((p[0] >> j) & 1u)
                | (((p[1] >> j) & 1u) << 1)
                | (((p[2] >> j) & 1u) << 2)
                | (((p[3] >> j) & 1u) << 3);
    unsigned i8 = i4
                | (((p[4] >> j) & 1u) << 4)
                | (((p[5] >> j) & 1u) << 5)
                | (((p[6] >> j) & 1u) << 6)
                | (((p[7] >> j) & 1u) << 7);
    ol[j] = (_Float16)l4[i4];
    oh[j] = (_Float16)l8[i8];
  }
}

// -------------------------- Pass 2: x f32 -> f16 ---------------------------
__global__ __launch_bounds__(256)
void xcast_kernel(const float* __restrict__ x, _Float16* __restrict__ xh) {
  size_t i = ((size_t)blockIdx.x * 256 + threadIdx.x) * 8;
  float4 a = *(const float4*)(x + i);
  float4 b = *(const float4*)(x + i + 4);
  union { _Float16 h[8]; uint4 u; } pk;
  pk.h[0] = (_Float16)a.x; pk.h[1] = (_Float16)a.y;
  pk.h[2] = (_Float16)a.z; pk.h[3] = (_Float16)a.w;
  pk.h[4] = (_Float16)b.x; pk.h[5] = (_Float16)b.y;
  pk.h[6] = (_Float16)b.z; pk.h[7] = (_Float16)b.w;
  *(uint4*)(xh + i) = pk.u;
}

// -------------------------- Pass 3: token mask -----------------------------
__global__ __launch_bounds__(256)
void token_mask_kernel(const float* __restrict__ x,
                       unsigned char* __restrict__ mask) {
  __shared__ float red[256];
  const int t   = threadIdx.x;
  const int tok = blockIdx.x;
  const float* xp = x + (size_t)tok * IN_F;
  float s = 0.0f;
  for (int i = t; i < IN_F; i += 256) { float v = xp[i]; s += v * v; }
  red[t] = s;
  __syncthreads();
  for (int off = 128; off > 0; off >>= 1) {
    if (t < off) red[t] += red[t + off];
    __syncthreads();
  }
  if (t == 0) mask[tok] = (sqrtf(red[0]) > 64.0f) ? 1 : 0;
}

// -------------------------- Pass 4: dual WMMA GEMM -------------------------
union FragH {   // one WMMA 16-bit operand (16 halves across 8 VGPRs per lane)
  v16h v;
  uint4 u[2];
};

// async-stage one 128x32 f16 tile (8 KB): 512 x 16B chunks over 256 threads
__device__ __forceinline__ void stage_tile(const _Float16* __restrict__ g,
                                           int grow0, int k0,
                                           _Float16 (*l)[BK], int tid) {
#pragma unroll
  for (int it = 0; it < 2; ++it) {
    int id = tid + it * 256;                 // 0..511
    int r = id >> 2;                         // 0..127
    int c = (id & 3) * 8;                    // 0,8,16,24
    async_copy_b128(g + (size_t)(grow0 + r) * IN_F + k0 + c, &l[r][c]);
  }
}

__global__ __launch_bounds__(256)
void gemm_dual_wmma_kernel(const _Float16* __restrict__ xh,
                           const _Float16* __restrict__ wl,
                           const _Float16* __restrict__ wh,
                           const unsigned char* __restrict__ mask,
                           const float* __restrict__ bias,
                           float* __restrict__ out) {
  __shared__ _Float16 As[2][BM][BK];   // 2 x 8 KB
  __shared__ _Float16 Bl[2][BN][BK];   // 2 x 8 KB
  __shared__ _Float16 Bh[2][BN][BK];   // 2 x 8 KB

  const int tid   = threadIdx.x;
  const int lane  = tid & 31;
  const int wave  = tid >> 5;          // 8 wave32 waves
  const int waveM = wave >> 1;         // 0..3 -> 32 token rows each
  const int waveN = wave & 1;          // 0..1 -> 64 output cols each
  const int m0 = blockIdx.y * BM;
  const int n0 = blockIdx.x * BN;

  v8f accL[2][4], accH[2][4];
  const v8f zero = {0.f, 0.f, 0.f, 0.f, 0.f, 0.f, 0.f, 0.f};
#pragma unroll
  for (int mt = 0; mt < 2; ++mt)
#pragma unroll
    for (int nt = 0; nt < 4; ++nt) { accL[mt][nt] = zero; accH[mt][nt] = zero; }

  // ISA 7.12.2 16-bit A/B lane mapping: lanes 0-15 take K[0..7]+K[16..23],
  // lanes 16-31 take K[8..15]+K[24..31]; row index = lane % 16.
  const int rsel = lane & 15;
  const int koff = (lane >> 4) << 3;   // 0 or 8

  // prologue: async-stage buffer 0
  stage_tile(xh, m0, 0, As[0], tid);
  stage_tile(wl, n0, 0, Bl[0], tid);
  stage_tile(wh, n0, 0, Bh[0], tid);
  wait_async_zero();
  __syncthreads();

  for (int k0 = 0; k0 < IN_F; k0 += BK) {
    const int buf = (k0 / BK) & 1;

    // issue async copies for the next K tile into the other buffer
    if (k0 + BK < IN_F) {
      stage_tile(xh, m0, k0 + BK, As[buf ^ 1], tid);
      stage_tile(wl, n0, k0 + BK, Bl[buf ^ 1], tid);
      stage_tile(wh, n0, k0 + BK, Bh[buf ^ 1], tid);
    }

    // --- A fragments for this wave's two 16-row tiles ---
    FragH fa[2];
#pragma unroll
    for (int mt = 0; mt < 2; ++mt) {
      const _Float16* p = &As[buf][waveM * 32 + mt * 16 + rsel][koff];
      fa[mt].u[0] = *(const uint4*)(p);
      fa[mt].u[1] = *(const uint4*)(p + 16);
    }

    // --- 4 N-tiles x 2 M-tiles x {lo,hi} = 16 WMMAs per K step ---
#pragma unroll
    for (int nt = 0; nt < 4; ++nt) {
      FragH fbl, fbh;
      const _Float16* pl = &Bl[buf][waveN * 64 + nt * 16 + rsel][koff];
      const _Float16* ph = &Bh[buf][waveN * 64 + nt * 16 + rsel][koff];
      fbl.u[0] = *(const uint4*)(pl);
      fbl.u[1] = *(const uint4*)(pl + 16);
      fbh.u[0] = *(const uint4*)(ph);
      fbh.u[1] = *(const uint4*)(ph + 16);
#pragma unroll
      for (int mt = 0; mt < 2; ++mt) {
        accL[mt][nt] = __builtin_amdgcn_wmma_f32_16x16x32_f16(
            false, fa[mt].v, false, fbl.v, (short)0, accL[mt][nt], false, false);
        accH[mt][nt] = __builtin_amdgcn_wmma_f32_16x16x32_f16(
            false, fa[mt].v, false, fbh.v, (short)0, accH[mt][nt], false, false);
      }
    }

    // my async copies done + everyone finished reading this buffer
    wait_async_zero();
    __syncthreads();
  }

  // --- epilogue: per-token precision select + bias ---
  // C/D layout (ISA 7.12.2): VGPR v, lanes 0-15 -> M=v, lanes 16-31 -> M=v+8;
  // N = lane % 16.
#pragma unroll
  for (int mt = 0; mt < 2; ++mt) {
    const int rbase = m0 + waveM * 32 + mt * 16 + ((lane >> 4) << 3);
#pragma unroll
    for (int nt = 0; nt < 4; ++nt) {
      const int col = n0 + waveN * 64 + nt * 16 + (lane & 15);
      const float bv = bias[col];
#pragma unroll
      for (int e = 0; e < 8; ++e) {
        const int row = rbase + e;
        const float v = mask[row] ? accH[mt][nt][e] : accL[mt][nt][e];
        out[(size_t)row * OUT_F + col] = v + bv;
      }
    }
  }
}

// ---------------------------------------------------------------------------
extern "C" void kernel_launch(void* const* d_in, const int* in_sizes, int n_in,
                              void* d_out, int out_size, void* d_ws, size_t ws_size,
                              hipStream_t stream) {
  const float* x    = (const float*)d_in[0];   // (4, 2048, 4096) f32
  const int*   q    = (const int*)  d_in[1];   // (8, 4096, 128) i32
  const float* lut4 = (const float*)d_in[2];   // (4096, 16) f32
  const float* lut8 = (const float*)d_in[3];   // (4096, 256) f32
  const float* bias = (const float*)d_in[4];   // (4096,) f32
  float* out = (float*)d_out;                  // (4, 2048, 4096) f32

  // workspace layout: [ W_lo f16 | W_hi f16 | x f16 | token mask u8 ] ~134 MB
  char* ws = (char*)d_ws;
  _Float16* wl = (_Float16*)(ws);
  _Float16* wh = (_Float16*)(ws + W_BYTES);
  _Float16* xh = (_Float16*)(ws + 2 * W_BYTES);
  unsigned char* mask = (unsigned char*)(ws + 2 * W_BYTES + XH_BYTES);

  dequant_kbit_kernel<<<(OUT_F * I32) / 256, 256, 0, stream>>>(q, lut4, lut8, wl, wh);
  xcast_kernel<<<(int)(((size_t)M_TOT * IN_F) / 8 / 256), 256, 0, stream>>>(x, xh);
  token_mask_kernel<<<M_TOT, 256, 0, stream>>>(x, mask);
  dim3 grid(OUT_F / BN, M_TOT / BM);           // (32, 64)
  gemm_dual_wmma_kernel<<<grid, 256, 0, stream>>>(xh, wl, wh, mask, bias, out);
}
